// AttentionalLSTM_62972810493995
// MI455X (gfx1250) — matrix operand order repaired and checked by reading, not compiled, not run on hardware
//
#include <hip/hip_runtime.h>
#include <hip/hip_bf16.h>
#include <math.h>

// Problem constants (from reference)
constexpr int kB  = 64;
constexpr int kT  = 128;
constexpr int kL  = 128;
constexpr int kH  = 1024;
constexpr int kIN = 1024;

typedef __attribute__((ext_vector_type(16))) __bf16 v16bf;
typedef __attribute__((ext_vector_type(8)))  float  v8f;

union AFrag {
  unsigned int d[8];   // 8 dwords = 16 bf16
  v16bf v;
};

__device__ __forceinline__ unsigned short f2bf(float f) {
  unsigned int u = __float_as_uint(f);
  unsigned int r = u + 0x7FFFu + ((u >> 16) & 1u);  // round-to-nearest-even
  return (unsigned short)(r >> 16);
}

__device__ __forceinline__ float sigmoidf_(float x) {
  return 1.0f / (1.0f + __expf(-x));
}

// A 16x32 bf16 fragment: V0..3 -> K pairs {0,1..6,7}+8*half ; V4..7 -> {16..23}+8*half
__device__ __forceinline__ void load_a(AFrag& a, const unsigned short* __restrict__ ar,
                                       int half) {
#pragma unroll
  for (int v = 0; v < 8; ++v) {
    const int pk = (v < 4) ? (2 * v) : (16 + 2 * (v - 4));
    a.d[v] = *(const unsigned int*)(ar + pk + 8 * half);
  }
}
// B 32x16 bf16 fragment: V0..3 -> K pairs {0,1..6,7}+8*half ; V4..7 -> {16..23}+8*half
__device__ __forceinline__ void load_b(AFrag& b, const unsigned short* __restrict__ br,
                                       int half) {
#pragma unroll
  for (int v = 0; v < 8; ++v)
    b.d[v] = *(const unsigned int*)(br + ((v < 4) ? (2 * v) : (16 + 2 * (v - 4))) + 8 * half);
}

// Software-pipelined K-segment (peeled epilogue, pointer-bump addressing):
// prefetch fragments for step s+1 while WMMA consumes step s.
template<int NT>
__device__ __forceinline__ void gemm_segment(v8f* __restrict__ c,
                                             const unsigned short* __restrict__ ar,
                                             const unsigned short* const* __restrict__ br,
                                             int Kseg, int half) {
  if (Kseg <= 0) return;
  AFrag a, b[NT];
  load_a(a, ar, half);
#pragma unroll
  for (int nt = 0; nt < NT; ++nt) load_b(b[nt], br[nt], half);

  const unsigned short* arp = ar + 32;
  const unsigned short* brp[NT];
#pragma unroll
  for (int nt = 0; nt < NT; ++nt) brp[nt] = br[nt] + 32;

  const int steps = Kseg >> 5;
  for (int s = 0; s < steps - 1; ++s) {
    AFrag an, bn[NT];
    load_a(an, arp, half);
#pragma unroll
    for (int nt = 0; nt < NT; ++nt) load_b(bn[nt], brp[nt], half);
#pragma unroll
    for (int nt = 0; nt < NT; ++nt)
      c[nt] = __builtin_amdgcn_wmma_f32_16x16x32_bf16(false, a.v, false, b[nt].v,
                                                      (short)0, c[nt], false, false);
    a = an;
#pragma unroll
    for (int nt = 0; nt < NT; ++nt) b[nt] = bn[nt];
    arp += 32;
#pragma unroll
    for (int nt = 0; nt < NT; ++nt) brp[nt] += 32;
  }
  // peeled last step: no prefetch
#pragma unroll
  for (int nt = 0; nt < NT; ++nt)
    c[nt] = __builtin_amdgcn_wmma_f32_16x16x32_bf16(false, a.v, false, b[nt].v,
                                                    (short)0, c[nt], false, false);
}

// ---------------------------------------------------------------------------
// Generic bf16 WMMA GEMM:  D[M,N] = act( A[M,K] * Bw[N,K]^T + bias(+bias2) + Cfull )
// One wave computes a 16 x (16*NT) tile.  amode selects A row addressing:
//   0: A + m*lda
//   1: A + ((m%B)*T + m/B)*lda          (input gathered as [T,B] rows)
//   2: k<Ka: A + (m*T + tpar)*lda ; k>=Ka: A2 + m*lda2   (fused [x_t|h], Bw2 likewise)
// ---------------------------------------------------------------------------
template<int NT>
__global__ void gemm_bf16_wmma(
    const unsigned short* __restrict__ A,
    const unsigned short* __restrict__ A2,
    const unsigned short* __restrict__ Bw,
    const unsigned short* __restrict__ Bw2,
    const float* __restrict__ bias,
    const float* __restrict__ bias2,
    const float* __restrict__ Cfull, int ldc,
    float* __restrict__ D, int ldd,
    unsigned short* __restrict__ Dbf, int lddbf,
    int M, int N, int K, int Ka,
    int lda, int lda2, int ldb, int ldb2,
    int amode, int tpar, int act)
{
  const int gw   = (blockIdx.x * blockDim.x + threadIdx.x) >> 5;
  const int lane = threadIdx.x & 31;
  const int tn_cnt = N / (16 * NT);
  const int total  = (M >> 4) * tn_cnt;
  if (gw >= total) return;                 // uniform per wave: EXEC stays all-ones
  const int tm = gw / tn_cnt;
  const int tn = gw - tm * tn_cnt;
  const int m0 = tm << 4, n0 = tn * (16 * NT);
  const int half = lane >> 4, lm = lane & 15;

  // ---- operand row pointers ----
  const int am = m0 + lm;                  // A lane->row: both halves M=0..15
  const unsigned short* arow0;
  if (amode == 1)      arow0 = A + (long)((am % kB) * kT + (am / kB)) * lda;
  else if (amode == 2) arow0 = A + (long)(am * kT + tpar) * lda;
  else                 arow0 = A + (long)am * lda;
  const unsigned short* brow0[NT];
#pragma unroll
  for (int nt = 0; nt < NT; ++nt) brow0[nt] = Bw + (long)(n0 + nt * 16 + lm) * ldb;

  // speculative L2/L0 prefetch of the first cachelines of each operand row
  // (lowers to global_prefetch_b8; no counters, graph-safe)
  __builtin_prefetch(arow0, 0, 1);
#pragma unroll
  for (int nt = 0; nt < NT; ++nt) __builtin_prefetch(brow0[nt], 0, 1);

  // ---- C init (f32 16x16 layout: VGPR r -> row m0+r+8*half, col +lm) ----
  v8f c[NT];
#pragma unroll
  for (int nt = 0; nt < NT; ++nt) {
    const int ccol = n0 + nt * 16 + lm;
    float bv = 0.0f;
    if (bias)  bv += bias[ccol];
    if (bias2) bv += bias2[ccol];
#pragma unroll
    for (int r = 0; r < 8; ++r) {
      float cv = bv;
      if (Cfull) cv += Cfull[(long)(m0 + r + 8 * half) * ldc + ccol];
      c[nt][r] = cv;
    }
  }

  // ---- segment 0: k in [0, min(Ka,K)) from A/Bw ----
  const int K0 = (Ka < K) ? Ka : K;
  gemm_segment<NT>(c, arow0, brow0, K0, half);

  // ---- segment 1 (fused [x|h] mode): k in [Ka, K) from A2/Bw2 ----
  if (K > K0 && A2 != nullptr) {
    const unsigned short* arow1 = A2 + (long)am * lda2;
    const unsigned short* brow1[NT];
#pragma unroll
    for (int nt = 0; nt < NT; ++nt) brow1[nt] = Bw2 + (long)(n0 + nt * 16 + lm) * ldb2;
    gemm_segment<NT>(c, arow1, brow1, K - K0, half);
  }

  // ---- epilogue ----
#pragma unroll
  for (int nt = 0; nt < NT; ++nt) {
    const int ccol = n0 + nt * 16 + lm;
#pragma unroll
    for (int r = 0; r < 8; ++r) {
      const int crow = m0 + r + 8 * half;
      float v = c[nt][r];
      if (act == 1) v = tanhf(v);
      if (D)   D[(long)crow * ldd + ccol] = v;
      if (Dbf) Dbf[(long)crow * lddbf + ccol] = f2bf(v);
    }
  }
}

// ---------------------------------------------------------------------------
// Elementwise helpers
// ---------------------------------------------------------------------------
__global__ void cvt_f32_bf16(const float* __restrict__ s,
                             unsigned short* __restrict__ d, int n) {
  int i = blockIdx.x * blockDim.x + threadIdx.x;
  if (i < n) d[i] = f2bf(s[i]);
}

__global__ void copy_f32(const float* __restrict__ s, float* __restrict__ d, int n) {
  int i = blockIdx.x * blockDim.x + threadIdx.x;
  if (i < n) d[i] = s[i];
}

// LSTM cell elementwise: gates [B,4H] (i,f,g,o), c in-place, hy -> out_raw (f32),
// hy_bf (for q GEMM), cat_bf[:,H:2H] (for output GEMM)
__global__ void lstm_ew(const float* __restrict__ gates,
                        float* __restrict__ c,
                        float* __restrict__ out_raw_t,
                        unsigned short* __restrict__ hy_bf,
                        unsigned short* __restrict__ cat_bf) {
  int i = blockIdx.x * blockDim.x + threadIdx.x;
  if (i >= kB * kH) return;
  int b = i / kH, h = i - b * kH;
  const float* g = gates + (long)b * 4 * kH;
  float ig = sigmoidf_(g[h]);
  float fg = sigmoidf_(g[kH + h]);
  float gg = tanhf(g[2 * kH + h]);
  float og = sigmoidf_(g[3 * kH + h]);
  float cy = fg * c[i] + ig * gg;
  float hy = og * tanhf(cy);
  c[i] = cy;
  out_raw_t[i] = hy;
  hy_bf[i] = f2bf(hy);
  cat_bf[(long)b * 2 * kH + kH + h] = f2bf(hy);
}

// scores[b,l] = dot(keys[b,l,:], q[b,:]) with mask -> -inf.  One wave per (b,l).
__global__ void scores_kernel(const float* __restrict__ keys,
                              const float* __restrict__ q,
                              const int* __restrict__ mask,
                              float* __restrict__ scores) {
  int w = (blockIdx.x * blockDim.x + threadIdx.x) >> 5;
  int lane = threadIdx.x & 31;
  if (w >= kB * kL) return;
  int b = w / kL;
  const float* kr = keys + (long)w * kH;
  const float* qr = q + (long)b * kH;
  float s = 0.0f;
  for (int h = lane; h < kH; h += 32) s += kr[h] * qr[h];
#pragma unroll
  for (int off = 16; off > 0; off >>= 1) s += __shfl_xor(s, off, 32);
  if (lane == 0) scores[w] = mask[w] ? -__builtin_huge_valf() : s;
}

// softmax over L=128 per batch row; writes output slice and ws probs
__global__ void softmax_kernel(const float* __restrict__ scores,
                               float* __restrict__ out_attn_t,
                               float* __restrict__ probs) {
  __shared__ float red[kL];
  int b = blockIdx.x, t = threadIdx.x;
  float s = scores[b * kL + t];
  red[t] = s; __syncthreads();
  for (int off = kL / 2; off > 0; off >>= 1) {
    if (t < off) red[t] = fmaxf(red[t], red[t + off]);
    __syncthreads();
  }
  float m = red[0]; __syncthreads();
  float e = __expf(s - m);
  red[t] = e; __syncthreads();
  for (int off = kL / 2; off > 0; off >>= 1) {
    if (t < off) red[t] += red[t + off];
    __syncthreads();
  }
  float p = e / red[0];
  out_attn_t[b * kL + t] = p;
  probs[b * kL + t] = p;
}

// ctx_w[b,h] = sum_l probs[b,l]*keys[b,l,h]; block = 256 h-lanes of one b.
__global__ void ctx_kernel(const float* __restrict__ keys,
                           const float* __restrict__ probs,
                           float* __restrict__ out_ctx_t,
                           unsigned short* __restrict__ cat_bf) {
  __shared__ float sp[kL];
  const int slabs = kH / 256;
  int b = blockIdx.x / slabs;
  int h = (blockIdx.x % slabs) * 256 + threadIdx.x;
  if (threadIdx.x < kL) sp[threadIdx.x] = probs[b * kL + threadIdx.x];
  __syncthreads();
  float acc = 0.0f;
  const float* kb = keys + (long)b * kL * kH + h;
#pragma unroll 4
  for (int l = 0; l < kL; ++l) acc += sp[l] * kb[(long)l * kH];
  out_ctx_t[(long)b * kH + h] = acc;
  cat_bf[(long)b * 2 * kH + h] = f2bf(acc);
}

// hT from strided output column T-1, cT from carry buffer
__global__ void final_copy(const float* __restrict__ out_output,
                           const float* __restrict__ cbuf,
                           float* __restrict__ hT, float* __restrict__ cT) {
  int i = blockIdx.x * blockDim.x + threadIdx.x;
  if (i >= kB * kH) return;
  int b = i / kH, h = i - b * kH;
  hT[i] = out_output[(long)b * kT * kH + (long)(kT - 1) * kH + h];
  cT[i] = cbuf[i];
}

// ---------------------------------------------------------------------------
extern "C" void kernel_launch(void* const* d_in, const int* in_sizes, int n_in,
                              void* d_out, int out_size, void* d_ws, size_t ws_size,
                              hipStream_t stream) {
  (void)in_sizes; (void)n_in; (void)out_size;
  const float* input  = (const float*)d_in[0];
  const float* h0     = (const float*)d_in[1];
  const float* c0     = (const float*)d_in[2];
  const float* ctxp   = (const float*)d_in[3];
  const int*   srcmask= (const int*)  d_in[4];
  const float* W_ih   = (const float*)d_in[5];
  const float* b_ih   = (const float*)d_in[6];
  const float* W_hh   = (const float*)d_in[7];
  const float* b_hh   = (const float*)d_in[8];
  const float* Wq     = (const float*)d_in[9];
  const float* bq     = (const float*)d_in[10];
  const float* Wk     = (const float*)d_in[11];
  const float* bk     = (const float*)d_in[12];
  const float* Wo     = (const float*)d_in[13];
  const float* bo     = (const float*)d_in[14];

  float* out        = (float*)d_out;
  float* out_output = out;                                  // [B,T,H]
  float* out_hT     = out_output + (size_t)kB * kT * kH;    // [B,H]
  float* out_cT     = out_hT + (size_t)kB * kH;             // [B,H]
  float* out_attn   = out_cT + (size_t)kB * kH;             // [T,B,L]
  float* out_ctxw   = out_attn + (size_t)kT * kB * kL;      // [T,B,H]
  float* out_raw    = out_ctxw + (size_t)kT * kB * kH;      // [T,B,H]

  // ---- carve workspace ----
  char* p = (char*)d_ws;
  auto carve = [&](size_t bytes) -> char* {
    char* r = p;
    p += (bytes + 255) & ~(size_t)255;
    return r;
  };
  unsigned short* in_bf  = (unsigned short*)carve((size_t)kB * kT * kIN * 2);
  unsigned short* ctx_bf = (unsigned short*)carve((size_t)kB * kL * kH * 2);
  unsigned short* Wih_bf = (unsigned short*)carve((size_t)4 * kH * kIN * 2);
  unsigned short* Whh_bf = (unsigned short*)carve((size_t)4 * kH * kH * 2);
  unsigned short* Wq_bf  = (unsigned short*)carve((size_t)kH * kH * 2);
  unsigned short* Wk_bf  = (unsigned short*)carve((size_t)kH * kH * 2);
  unsigned short* Wo_bf  = (unsigned short*)carve((size_t)kH * 2 * kH * 2);
  float* keys    = (float*)carve((size_t)kB * kL * kH * 4);
  float* gates   = (float*)carve((size_t)kB * 4 * kH * 4);
  float* cbuf    = (float*)carve((size_t)kB * kH * 4);
  unsigned short* h_bf  = (unsigned short*)carve((size_t)kB * kH * 2);
  unsigned short* hy_bf = (unsigned short*)carve((size_t)kB * kH * 2);
  unsigned short* cat_bf= (unsigned short*)carve((size_t)kB * 2 * kH * 2);
  float* qbuf    = (float*)carve((size_t)kB * kH * 4);
  float* scores  = (float*)carve((size_t)kB * kL * 4);
  float* probs   = (float*)carve((size_t)kB * kL * 4);
  size_t base_used = (size_t)(p - (char*)d_ws);
  const size_t xproj_bytes = (size_t)kT * kB * 4 * kH * 4;   // 128 MB
  float* xproj = nullptr;
  bool precompute = (ws_size >= base_used + xproj_bytes);
  if (precompute) xproj = (float*)carve(xproj_bytes);

  auto cv = [&](const float* s, unsigned short* d, int n) {
    cvt_f32_bf16<<<(n + 255) / 256, 256, 0, stream>>>(s, d, n);
  };
  auto gemm = [&](const unsigned short* A, const unsigned short* A2,
                  const unsigned short* Bw, const unsigned short* Bw2,
                  const float* bias, const float* bias2,
                  const float* Cfull, int ldc,
                  float* D, int ldd, unsigned short* Dbf, int lddbf,
                  int M, int N, int K, int Ka,
                  int lda, int lda2, int ldb, int ldb2,
                  int amode, int tpar, int act, int ntile) {
    int waves = (M / 16) * (N / (16 * ntile));
    int blocks = (waves + 7) / 8;
    switch (ntile) {
      case 1:
        gemm_bf16_wmma<1><<<blocks, 256, 0, stream>>>(A, A2, Bw, Bw2, bias, bias2,
            Cfull, ldc, D, ldd, Dbf, lddbf, M, N, K, Ka, lda, lda2, ldb, ldb2,
            amode, tpar, act);
        break;
      case 2:
        gemm_bf16_wmma<2><<<blocks, 256, 0, stream>>>(A, A2, Bw, Bw2, bias, bias2,
            Cfull, ldc, D, ldd, Dbf, lddbf, M, N, K, Ka, lda, lda2, ldb, ldb2,
            amode, tpar, act);
        break;
      default:
        gemm_bf16_wmma<4><<<blocks, 256, 0, stream>>>(A, A2, Bw, Bw2, bias, bias2,
            Cfull, ldc, D, ldd, Dbf, lddbf, M, N, K, Ka, lda, lda2, ldb, ldb2,
            amode, tpar, act);
        break;
    }
  };

  // ---- one-time conversions ----
  cv(input, in_bf, kB * kT * kIN);
  cv(ctxp, ctx_bf, kB * kL * kH);
  cv(W_ih, Wih_bf, 4 * kH * kIN);
  cv(W_hh, Whh_bf, 4 * kH * kH);
  cv(Wq, Wq_bf, kH * kH);
  cv(Wk, Wk_bf, kH * kH);
  cv(Wo, Wo_bf, kH * 2 * kH);
  cv(h0, h_bf, kB * kH);
  copy_f32<<<(kB * kH + 255) / 256, 256, 0, stream>>>(c0, cbuf, kB * kH);

  // ---- time-invariant keys projection: att_keys = ctx @ Wk^T + bk ----
  gemm(ctx_bf, nullptr, Wk_bf, nullptr, bk, nullptr, nullptr, 0,
       keys, kH, nullptr, 0,
       kB * kL, kH, kH, kH, kH, 0, kH, 0, /*amode*/0, 0, /*act*/0, /*NT*/4);

  // ---- hoisted input projection: xproj[t,b,:] = x_t @ W_ih^T + b_ih + b_hh ----
  if (precompute) {
    gemm(in_bf, nullptr, Wih_bf, nullptr, b_ih, b_hh, nullptr, 0,
         xproj, 4 * kH, nullptr, 0,
         kB * kT, 4 * kH, kIN, kIN, kIN, 0, kIN, 0, /*amode*/1, 0, /*act*/0, /*NT*/4);
  }

  // ---- sequential recurrence ----
  for (int t = 0; t < kT; ++t) {
    if (precompute) {
      // gates = h @ W_hh^T + xproj[t]
      gemm(h_bf, nullptr, Whh_bf, nullptr, nullptr, nullptr,
           xproj + (size_t)t * kB * 4 * kH, 4 * kH,
           gates, 4 * kH, nullptr, 0,
           kB, 4 * kH, kH, kH, kH, 0, kH, 0, /*amode*/0, 0, 0, /*NT*/2);
    } else {
      // gates = [x_t | h] @ [W_ih ; W_hh]^T + b_ih + b_hh
      gemm(in_bf, h_bf, Wih_bf, Whh_bf, b_ih, b_hh, nullptr, 0,
           gates, 4 * kH, nullptr, 0,
           kB, 4 * kH, kIN + kH, kIN, kIN, kH, kIN, kH, /*amode*/2, t, 0, /*NT*/2);
    }
    lstm_ew<<<(kB * kH + 255) / 256, 256, 0, stream>>>(
        gates, cbuf, out_raw + (size_t)t * kB * kH, hy_bf, cat_bf);
    // q = hy @ Wq^T + bq
    gemm(hy_bf, nullptr, Wq_bf, nullptr, bq, nullptr, nullptr, 0,
         qbuf, kH, nullptr, 0,
         kB, kH, kH, kH, kH, 0, kH, 0, 0, 0, 0, /*NT*/1);
    scores_kernel<<<(kB * kL) / 8, 256, 0, stream>>>(keys, qbuf, srcmask, scores);
    softmax_kernel<<<kB, kL, 0, stream>>>(scores, out_attn + (size_t)t * kB * kL, probs);
    ctx_kernel<<<kB * (kH / 256), 256, 0, stream>>>(
        keys, probs, out_ctxw + (size_t)t * kB * kH, cat_bf);
    // h_tilde = tanh([ctx_w|hy] @ Wo^T + bo) -> output[b,t,:] (strided) + bf16 carry
    gemm(cat_bf, nullptr, Wo_bf, nullptr, bo, nullptr, nullptr, 0,
         out_output + (size_t)t * kH, kT * kH, h_bf, kH,
         kB, kH, 2 * kH, 2 * kH, 2 * kH, 0, 2 * kH, 0, 0, 0, /*act=tanh*/1, /*NT*/1);
  }

  final_copy<<<(kB * kH + 255) / 256, 256, 0, stream>>>(out_output, cbuf, out_hT, out_cT);
}
